// CSABlock_11115375362189
// MI455X (gfx1250) — compile-verified, matrix operand order checked by reading, not run on hardware
//
#include <hip/hip_runtime.h>

#define N_   4
#define C_   256
#define IC_  128
#define D_   9
#define HW_  4096
#define NCHW_  (N_*C_*HW_)    // 4,194,304
#define NHWIC_ (N_*HW_*IC_)   // 2,097,152

typedef __attribute__((ext_vector_type(16))) __bf16 v16bf;
typedef __attribute__((ext_vector_type(8)))  float  v8f;

static __device__ __forceinline__ v8f vzero8() {
  v8f z = {0.f,0.f,0.f,0.f,0.f,0.f,0.f,0.f};
  return z;
}

static __device__ __forceinline__ v8f wmma_bf16(v16bf a, v16bf b, v8f c) {
  return __builtin_amdgcn_wmma_f32_16x16x32_bf16(false, a, false, b, (short)0, c, false, false);
}

// 16x32 bf16 fragment from a [row][K] tile (row stride ld elements, K contiguous).
// Per ISA 7.12.2: lanes 0-15 => M=lane, K {0..7,16..23}; lanes 16-31 => K {8..15,24..31}.
// Two contiguous 16B chunks per lane -> ds_load_b128 / global_load_b128 when
// base and ld are 16B-aligned (ld must be a multiple of 8 elements).
static __device__ __forceinline__ v16bf load_frag(const __bf16* base, int ld, int lane) {
  const int half = lane >> 4;
  const int m    = lane & 15;
  const __bf16* r = base + m * ld + half * 8;
  union { uint4 u[2]; v16bf v; } u;
  u.u[0] = *(const uint4*)r;
  u.u[1] = *(const uint4*)(r + 16);
  return u.v;
}

// Same fragment gather from a row-major f32 matrix (weights), converting to bf16.
static __device__ __forceinline__ v16bf load_frag_f32(const float* base, int ld, int lane) {
  const int half = lane >> 4;
  const int m    = lane & 15;
  const float* r = base + m * ld + half * 8;
  v16bf f;
#pragma unroll
  for (int v = 0; v < 4; ++v) {
    f[2*v]     = (__bf16)r[2*v];
    f[2*v + 1] = (__bf16)r[2*v + 1];
    f[8 + 2*v]     = (__bf16)r[16 + 2*v];
    f[8 + 2*v + 1] = (__bf16)r[16 + 2*v + 1];
  }
  return f;
}

// Async global->LDS copy of one 16B chunk (GLOBAL_LOAD_ASYNC_TO_LDS_B128,
// ASYNCcnt-tracked).  Low 32 bits of a generic LDS pointer are the raw LDS
// byte offset (flat aperture rule), which is what VDST wants.
static __device__ __forceinline__ void async_g2l_b128(__bf16* lds_dst, const __bf16* gsrc) {
  unsigned ldsoff = (unsigned)(unsigned long long)lds_dst;
  asm volatile("global_load_async_to_lds_b128 %0, %1, off"
               :: "v"(ldsoff), "v"(gsrc) : "memory");
}
static __device__ __forceinline__ void wait_async0() {
  asm volatile("s_wait_asynccnt 0x0" ::: "memory");
}

// ---------------------------------------------------------------------------
// Kernel 1: center slice + max over depth.  Pure bandwidth, float4 vectorized.
// ---------------------------------------------------------------------------
__global__ __launch_bounds__(256) void k_reduce(const float* __restrict__ feat,
                                                float* __restrict__ center,
                                                float* __restrict__ xmax) {
  const int idx = blockIdx.x * 256 + threadIdx.x;      // over N*C*HW/4
  const int hw4 = idx & (HW_/4 - 1);
  const int nc  = idx >> 10;                            // HW_/4 == 1024
  const float4* base = (const float4*)feat + (size_t)nc * (D_ * HW_ / 4) + hw4;
  float4 m = base[0];
  float4 cen = m;
#pragma unroll
  for (int d = 1; d < D_; ++d) {
    float4 v = base[(size_t)d * (HW_/4)];
    if (d == D_/2 + 1) cen = v;
    m.x = fmaxf(m.x, v.x); m.y = fmaxf(m.y, v.y);
    m.z = fmaxf(m.z, v.z); m.w = fmaxf(m.w, v.w);
  }
  ((float4*)center)[idx] = cen;
  ((float4*)xmax)[idx]   = m;
}

// ---------------------------------------------------------------------------
// Kernel 2: theta/phi/g = conv1x1 (+BN+ReLU).  theta/phi stored (n,hw,ic) bf16;
// g stored (n,ic,hw) bf16 so the attention kernel can async-stage it directly
// as B fragments (K contiguous) with no transpose.
// Tile row stride 56 elem = 112B = 7*16B: b128-legal and bank-conflict-free.
// ---------------------------------------------------------------------------
#define LQ 56
__global__ __launch_bounds__(256) void k_qkv(
    const float* __restrict__ center, const float* __restrict__ xmax,
    const float* __restrict__ w_theta, const float* __restrict__ w_phi,
    const float* __restrict__ w_g,
    const float* __restrict__ th_g, const float* __restrict__ th_b,
    const float* __restrict__ th_m, const float* __restrict__ th_v,
    const float* __restrict__ ph_g, const float* __restrict__ ph_b,
    const float* __restrict__ ph_m, const float* __restrict__ ph_v,
    __bf16* __restrict__ theta_t, __bf16* __restrict__ phi_t,
    __bf16* __restrict__ g_c) {
  __shared__ __align__(16) __bf16 sAc[128 * LQ];    // center tile [hw][c]
  __shared__ __align__(16) __bf16 sAx[128 * LQ];    // xmax tile   [hw][c]
  __shared__ __align__(16) __bf16 sW[3][128 * LQ];  // weights     [ic][c]

  const int tid  = threadIdx.x;
  const int lane = tid & 31;
  const int wid  = tid >> 5;
  const int n    = blockIdx.x >> 5;
  const int q0   = (blockIdx.x & 31) * 128;

  v8f a_th[8], a_ph[8], a_g[8];
#pragma unroll
  for (int i = 0; i < 8; ++i) { a_th[i] = vzero8(); a_ph[i] = vzero8(); a_g[i] = vzero8(); }

  for (int cc = 0; cc < C_; cc += 32) {
    // stage A tiles: pack bf16 pairs (two c rows) -> b32 LDS stores
    {
      const int ci2 = tid >> 4;           // 0..15 -> c pair
      const int hwc = (tid & 15) * 8;     // 0..120
      const float* p0c = center + (size_t)(n * C_ + cc + ci2 * 2) * HW_ + q0 + hwc;
      const float* p1c = p0c + HW_;
      const float* p0x = xmax   + (size_t)(n * C_ + cc + ci2 * 2) * HW_ + q0 + hwc;
      const float* p1x = p0x + HW_;
#pragma unroll
      for (int j = 0; j < 8; ++j) {
        union { __bf16 h[2]; unsigned w; } pc, px;
        pc.h[0] = (__bf16)p0c[j]; pc.h[1] = (__bf16)p1c[j];
        px.h[0] = (__bf16)p0x[j]; px.h[1] = (__bf16)p1x[j];
        *(unsigned*)&sAc[(hwc + j) * LQ + ci2 * 2] = pc.w;
        *(unsigned*)&sAx[(hwc + j) * LQ + ci2 * 2] = px.w;
      }
    }
    // stage weight tiles [ic][32c]: pack 8 bf16 -> b128 LDS stores
    {
      const int ic = tid >> 1;            // 0..127
      const int c8 = (tid & 1) * 16;      // 0 or 16
      const float* pt = w_theta + (size_t)ic * C_ + cc + c8;
      const float* pp = w_phi   + (size_t)ic * C_ + cc + c8;
      const float* pg = w_g     + (size_t)ic * C_ + cc + c8;
#pragma unroll
      for (int seg = 0; seg < 2; ++seg) {
        union { __bf16 h[8]; uint4 q; } ut, up, ug;
#pragma unroll
        for (int j = 0; j < 8; ++j) {
          ut.h[j] = (__bf16)pt[seg * 8 + j];
          up.h[j] = (__bf16)pp[seg * 8 + j];
          ug.h[j] = (__bf16)pg[seg * 8 + j];
        }
        *(uint4*)&sW[0][ic * LQ + c8 + seg * 8] = ut.q;
        *(uint4*)&sW[1][ic * LQ + c8 + seg * 8] = up.q;
        *(uint4*)&sW[2][ic * LQ + c8 + seg * 8] = ug.q;
      }
    }
    __syncthreads();

    v16bf Ac = load_frag(sAc + (wid * 16) * LQ, LQ, lane);
    v16bf Ax = load_frag(sAx + (wid * 16) * LQ, LQ, lane);
#pragma unroll
    for (int ot = 0; ot < 8; ++ot) {
      v16bf Bt = load_frag(sW[0] + (ot * 16) * LQ, LQ, lane);
      v16bf Bp = load_frag(sW[1] + (ot * 16) * LQ, LQ, lane);
      v16bf Bg = load_frag(sW[2] + (ot * 16) * LQ, LQ, lane);
      a_th[ot] = wmma_bf16(Ac, Bt, a_th[ot]);
      a_ph[ot] = wmma_bf16(Ax, Bp, a_ph[ot]);
      a_g[ot]  = wmma_bf16(Ax, Bg, a_g[ot]);
    }
    __syncthreads();
  }

  // BN + ReLU fused epilogue
  const int half = lane >> 4;
  const int nl   = lane & 15;
#pragma unroll
  for (int ot = 0; ot < 8; ++ot) {
    const int ic = ot * 16 + nl;
    const float ts = th_g[ic] * rsqrtf(th_v[ic] + 1e-5f);
    const float tm = th_m[ic], tb = th_b[ic];
    const float ps = ph_g[ic] * rsqrtf(ph_v[ic] + 1e-5f);
    const float pm = ph_m[ic], pb = ph_b[ic];
#pragma unroll
    for (int i = 0; i < 8; ++i) {
      const int q = q0 + wid * 16 + i + 8 * half;
      const size_t o = (size_t)(n * HW_ + q) * IC_ + ic;       // (n,hw,ic)
      theta_t[o] = (__bf16)fmaxf((a_th[ot][i] - tm) * ts + tb, 0.f);
      phi_t[o]   = (__bf16)fmaxf((a_ph[ot][i] - pm) * ps + pb, 0.f);
      g_c[(size_t)(n * IC_ + ic) * HW_ + q] = (__bf16)a_g[ot][i];  // (n,ic,hw)
    }
  }
}

// ---------------------------------------------------------------------------
// Kernel 3: flash attention, double-buffered async staging.
// theta A-fragments live in VGPRs for the whole kernel (global b128 loads).
// phi [k][c] and g [ic][k] tiles stream via GLOBAL_LOAD_ASYNC_TO_LDS_B128
// into ping-pong buffers; s_wait_asynccnt 0 only at iteration end, so the
// copy of tile t+1 overlaps all of tile t's WMMA + softmax work.
// ---------------------------------------------------------------------------
#define LT 136
#define LK 72
__global__ __launch_bounds__(256) void k_attn(
    const __bf16* __restrict__ theta_t, const __bf16* __restrict__ phi_t,
    const __bf16* __restrict__ g_c, __bf16* __restrict__ obuf) {
  __shared__ __align__(16) __bf16 sPh[2][64 * LT];   // phi tile [k][c]
  __shared__ __align__(16) __bf16 sGt[2][128 * LK];  // g tile   [ic][k]
  __shared__ __align__(16) float  sS[128 * 66];      // scores   [q][k]
  __shared__ __align__(16) __bf16 sP[128 * LK];      // exp(scores)
  __shared__ float  sM[128], sL[128], sScale[128];

  const int tid  = threadIdx.x;
  const int lane = tid & 31;
  const int wid  = tid >> 5;
  const int n    = blockIdx.x >> 5;
  const int q0   = (blockIdx.x & 31) * 128;
  const int half = lane >> 4;
  const int nl   = lane & 15;

  // theta A-fragments: 16 q rows per wave, K=128 -> 4 resident fragments
  v16bf thA[4];
#pragma unroll
  for (int cb = 0; cb < 4; ++cb)
    thA[cb] = load_frag(theta_t + (size_t)(n * HW_ + q0 + wid * 16) * IC_ + cb * 32,
                        IC_, lane);

  if (tid < 128) { sM[tid] = -3.0e38f; sL[tid] = 0.f; }

  v8f O[8];
#pragma unroll
  for (int i = 0; i < 8; ++i) O[i] = vzero8();

  // stage tile 0 into buffer 0
#pragma unroll
  for (int i = 0; i < 4; ++i) {
    const int e = tid + i * 256;
    const int r = e >> 4, ch = e & 15;            // phi: 64 rows x 16 chunks
    async_g2l_b128(&sPh[0][r * LT + ch * 8],
                   phi_t + (size_t)(n * HW_ + r) * IC_ + ch * 8);
    const int ic = e >> 3, ck = e & 7;            // g: 128 rows x 8 chunks
    async_g2l_b128(&sGt[0][ic * LK + ck * 8],
                   g_c + (size_t)(n * IC_ + ic) * HW_ + ck * 8);
  }
  wait_async0();
  __syncthreads();

  const int nIter = HW_ / 64;
  for (int t = 0; t < nIter; ++t) {
    const int buf = t & 1;
    // kick off async staging of tile t+1 into the other buffer
    if (t + 1 < nIter) {
      const int k1 = (t + 1) * 64;
#pragma unroll
      for (int i = 0; i < 4; ++i) {
        const int e = tid + i * 256;
        const int r = e >> 4, ch = e & 15;
        async_g2l_b128(&sPh[buf ^ 1][r * LT + ch * 8],
                       phi_t + (size_t)(n * HW_ + k1 + r) * IC_ + ch * 8);
        const int ic = e >> 3, ck = e & 7;
        async_g2l_b128(&sGt[buf ^ 1][ic * LK + ck * 8],
                       g_c + (size_t)(n * IC_ + ic) * HW_ + k1 + ck * 8);
      }
    }

    // S = theta @ phi^T : 4 16x16 n-tiles, K=128 in 4 WMMA steps each
    v8f s[4];
#pragma unroll
    for (int kt = 0; kt < 4; ++kt) s[kt] = vzero8();
#pragma unroll
    for (int cb = 0; cb < 4; ++cb) {
#pragma unroll
      for (int kt = 0; kt < 4; ++kt) {
        v16bf b = load_frag(&sPh[buf][(kt * 16) * LT + cb * 32], LT, lane);
        s[kt] = wmma_bf16(thA[cb], b, s[kt]);
      }
    }
#pragma unroll
    for (int kt = 0; kt < 4; ++kt)
#pragma unroll
      for (int i = 0; i < 8; ++i)
        sS[(wid * 16 + i + 8 * half) * 66 + kt * 16 + nl] = s[kt][i];
    __syncthreads();

    // online softmax: 2 threads per q row, pair-combine via lane shuffles
    {
      const int r = tid >> 1;
      const int h = tid & 1;
      const float* row = sS + r * 66 + h * 32;
      float mp = -3.0e38f;
#pragma unroll 8
      for (int j = 0; j < 32; ++j) mp = fmaxf(mp, row[j]);
      mp = fmaxf(mp, __shfl_xor(mp, 1));
      const float mOld = sM[r];
      const float mNew = fmaxf(mOld, mp);
      const float corr = __expf(mOld - mNew);
      float lsum = 0.f;
      __bf16* pr = sP + r * LK + h * 32;
#pragma unroll 8
      for (int j = 0; j < 32; ++j) {
        const float e = __expf(row[j] - mNew);
        lsum += e;
        pr[j] = (__bf16)e;
      }
      lsum += __shfl_xor(lsum, 1);
      if (h == 0) { sM[r] = mNew; sL[r] = sL[r] * corr + lsum; sScale[r] = corr; }
    }
    __syncthreads();

    // O = O * corr + P @ g^T
#pragma unroll
    for (int i = 0; i < 8; ++i) {
      const float sc = sScale[wid * 16 + i + 8 * half];
#pragma unroll
      for (int it = 0; it < 8; ++it) O[it][i] *= sc;
    }
#pragma unroll
    for (int kb = 0; kb < 2; ++kb) {
      v16bf a = load_frag(sP + (wid * 16) * LK + kb * 32, LK, lane);
#pragma unroll
      for (int it = 0; it < 8; ++it) {
        v16bf b = load_frag(&sGt[buf][(it * 16) * LK + kb * 32], LK, lane);
        O[it] = wmma_bf16(a, b, O[it]);
      }
    }
    wait_async0();     // tile t+1 resident before anyone flips buffers
    __syncthreads();
  }

  // normalize and store bf16 (n, hw, ic)
#pragma unroll
  for (int i = 0; i < 8; ++i) {
    const int q = q0 + wid * 16 + i + 8 * half;
    const float inv = 1.f / sL[wid * 16 + i + 8 * half];
#pragma unroll
    for (int it = 0; it < 8; ++it)
      obuf[(size_t)(n * HW_ + q) * IC_ + it * 16 + nl] = (__bf16)(O[it][i] * inv);
  }
}

// ---------------------------------------------------------------------------
// Kernel 4: out = conv1x1(weighted, w_w) + center_x.  WMMA GEMM
// (128 hw x 128 ic) x (128 ic x 256 o) with residual epilogue.
// ---------------------------------------------------------------------------
__global__ __launch_bounds__(256) void k_out(
    const __bf16* __restrict__ obuf, const float* __restrict__ w_w,
    const float* __restrict__ center, float* __restrict__ out) {
  const int tid  = threadIdx.x;
  const int lane = tid & 31;
  const int wid  = tid >> 5;
  const int n    = blockIdx.x >> 5;
  const int q0   = (blockIdx.x & 31) * 128;
  const int half = lane >> 4;
  const int nl   = lane & 15;

  v8f acc[16];
#pragma unroll
  for (int i = 0; i < 16; ++i) acc[i] = vzero8();

#pragma unroll
  for (int kb = 0; kb < IC_; kb += 32) {
    v16bf a = load_frag(obuf + (size_t)(n * HW_ + q0 + wid * 16) * IC_ + kb, IC_, lane);
#pragma unroll
    for (int ot = 0; ot < 16; ++ot) {
      v16bf b = load_frag_f32(w_w + (size_t)(ot * 16) * IC_ + kb, IC_, lane);
      acc[ot] = wmma_bf16(a, b, acc[ot]);
    }
  }
#pragma unroll
  for (int ot = 0; ot < 16; ++ot) {
#pragma unroll
    for (int i = 0; i < 8; ++i) {
      const int q = q0 + wid * 16 + i + 8 * half;
      const int o = ot * 16 + nl;
      const size_t idx = (size_t)(n * C_ + o) * HW_ + q;
      out[idx] = acc[ot][i] + center[idx];
    }
  }
}

// ---------------------------------------------------------------------------
extern "C" void kernel_launch(void* const* d_in, const int* in_sizes, int n_in,
                              void* d_out, int out_size, void* d_ws, size_t ws_size,
                              hipStream_t stream) {
  const float* feature = (const float*)d_in[0];
  const float* w_theta = (const float*)d_in[1];
  const float* th_g    = (const float*)d_in[2];
  const float* th_b    = (const float*)d_in[3];
  const float* th_m    = (const float*)d_in[4];
  const float* th_v    = (const float*)d_in[5];
  const float* w_phi   = (const float*)d_in[6];
  const float* ph_g    = (const float*)d_in[7];
  const float* ph_b    = (const float*)d_in[8];
  const float* ph_m    = (const float*)d_in[9];
  const float* ph_v    = (const float*)d_in[10];
  const float* w_g     = (const float*)d_in[11];
  const float* w_w     = (const float*)d_in[12];

  float* wsf    = (float*)d_ws;
  float* center = wsf;                 // N*C*HW f32
  float* xmax   = wsf + NCHW_;         // N*C*HW f32
  __bf16* bb      = (__bf16*)(wsf + 2 * (size_t)NCHW_);
  __bf16* theta_t = bb;                // (n,hw,ic) bf16
  __bf16* phi_t   = bb + (size_t)NHWIC_;
  __bf16* g_c     = bb + 2 * (size_t)NHWIC_;   // (n,ic,hw) bf16
  __bf16* obuf    = bb + 3 * (size_t)NHWIC_;   // (n,hw,ic) bf16

  k_reduce<<<NCHW_ / 4 / 256, 256, 0, stream>>>(feature, center, xmax);
  k_qkv<<<N_ * (HW_ / 128), 256, 0, stream>>>(center, xmax, w_theta, w_phi, w_g,
                                              th_g, th_b, th_m, th_v,
                                              ph_g, ph_b, ph_m, ph_v,
                                              theta_t, phi_t, g_c);
  k_attn<<<N_ * (HW_ / 128), 256, 0, stream>>>(theta_t, phi_t, g_c, obuf);
  k_out<<<N_ * (HW_ / 128), 256, 0, stream>>>(obuf, w_w, center, (float*)d_out);
}
